// MultiHeadAttention_21595095564443
// MI455X (gfx1250) — matrix operand order repaired and checked by reading, not compiled, not run on hardware
//
#include <hip/hip_runtime.h>
#include <hip/hip_bf16.h>

// ---------------------------------------------------------------------------
// MI455X (gfx1250) fused pre-LN multi-head attention block.
// ~1300 FLOP/byte -> compute bound: all matmuls via v_wmma_f32_16x16x32_bf16
// (fp32 accumulate). GEMMs use 64x64-per-wave tiles (1 b128 load per WMMA).
// Attention stages K / V^T chunks into LDS with the Tensor Data Mover
// (tensor_load_to_lds, TENSORcnt, double-buffered) when available.
// ---------------------------------------------------------------------------

#define EMBED 768
#define SEQ 2048
#define BATCH 2
#define HEADS 12
#define HDIM 64
#define THREE_D 2304
#define MTOK 4096 /* BATCH*SEQ */

typedef __attribute__((ext_vector_type(16))) __bf16 v16bf;
typedef __attribute__((ext_vector_type(8))) float v8f;
typedef __attribute__((ext_vector_type(4))) unsigned int v4u;
typedef __attribute__((ext_vector_type(8))) int v8i;
typedef __attribute__((ext_vector_type(4))) int v4i;

#if defined(__HIP_DEVICE_COMPILE__) && __has_builtin(__builtin_amdgcn_tensor_load_to_lds) && \
    __has_builtin(__builtin_amdgcn_s_wait_tensorcnt)
#define USE_TDM 1
#else
#define USE_TDM 0
#endif

union FragU {
    v16bf v;
    uint4 q[2];
};

// round-to-nearest-even float -> bf16
static __device__ inline __bf16 f2bf(float f) {
    unsigned int u = __float_as_uint(f);
    u += 0x7FFFu + ((u >> 16) & 1u);
    unsigned short s = (unsigned short)(u >> 16);
    __bf16 r;
    __builtin_memcpy(&r, &s, 2);
    return r;
}

// A-matrix 16x32 bf16 fragment (ISA 7.12.2): lanes 0-15 hold K={0..7,16..23},
// lanes 16-31 hold K={8..15,24..31}; rowp = &A[row][kb] for this lane's row.
static __device__ inline v16bf load_frag_a(const __bf16* rowp, int lane) {
    FragU f;
    const int koff = (lane & 16) ? 8 : 0;
    f.q[0] = *(const uint4*)(rowp + koff);
    f.q[1] = *(const uint4*)(rowp + koff + 16);
    return f.v;
}

// B-matrix 32x16 bf16 fragment: lane n=lane&15, same K order as A:
// lanes 0-15 -> K={0..7,16..23}, lanes 16-31 -> K={8..15,24..31}; colp = &Bt[n][kb].
static __device__ inline v16bf load_frag_b(const __bf16* colp, int lane) {
    FragU f;
    const int kst = (lane & 16) ? 8 : 0;
    f.q[0] = *(const uint4*)(colp + kst);
    f.q[1] = *(const uint4*)(colp + kst + 16);
    return f.v;
}

static __device__ inline v8f wmma_bf16(v16bf a, v16bf b, v8f c) {
    // 8 args: (neg_a, A, neg_b, B, c_mod, C, reuse_a, reuse_b)
    return __builtin_amdgcn_wmma_f32_16x16x32_bf16(false, a, false, b, (short)0, c,
                                                   false, false);
}

#if USE_TDM
// Issue a 2D TDM load: tile_d1 rows of tile_d0 bf16 elements (row stride
// `stride0` elements) from `gptr` into LDS byte offset `lds_off`.
// D# layout per CDNA5 ISA sections 8.3/8.4 (data_size=1 -> 2 bytes, type=2).
static __device__ inline void tdm_load_2d(unsigned lds_off, const void* gptr,
                                          unsigned tile_d0, unsigned tile_d1,
                                          unsigned long long stride0) {
    const unsigned long long ga = (unsigned long long)(size_t)gptr;
    v4u g0;
    g0[0] = 1u;                                    // count=1 (valid user D#)
    g0[1] = lds_off;                               // lds_addr [63:32]
    g0[2] = (unsigned)(ga & 0xFFFFFFFFu);          // global_addr [95:64]
    g0[3] = (unsigned)((ga >> 32) & 0x1FFFFFFu)    // global_addr [120:96]
            | (2u << 30);                          // type=2 ("image") [127:126]
    v8i g1;
    g1[0] = (int)(1u << 16);                       // data_size=1 (2B); wg_mask=0
    g1[1] = (int)((tile_d0 & 0xFFFFu) << 16);      // tensor_dim0[15:0] @ [63:48]
    g1[2] = (int)(((tile_d0 >> 16) & 0xFFFFu) |    // tensor_dim0[31:16]
                  ((tile_d1 & 0xFFFFu) << 16));    // tensor_dim1[15:0]
    g1[3] = (int)(((tile_d1 >> 16) & 0xFFFFu) |    // tensor_dim1[31:16]
                  ((tile_d0 & 0xFFFFu) << 16));    // tile_dim0 @ [127:112]
    g1[4] = (int)(tile_d1 & 0xFFFFu);              // tile_dim1; tile_dim2=0
    g1[5] = (int)(unsigned)(stride0 & 0xFFFFFFFFull);        // dim0 stride lo
    g1[6] = (int)(unsigned)((stride0 >> 32) & 0xFFFFull);    // stride hi; dim1 stride=0
    g1[7] = 0;
    const v4i z = {0, 0, 0, 0};
#if __clang_major__ >= 23
    const v8i z8 = {0, 0, 0, 0, 0, 0, 0, 0};
    __builtin_amdgcn_tensor_load_to_lds(g0, g1, z, z, z8, 0);
#else
    __builtin_amdgcn_tensor_load_to_lds(g0, g1, z, z, 0);
#endif
}
#endif

// ---------------------------------------------------------------------------
// Kernel 1: LayerNorm, one wave per token; 24 elements per lane.
// ---------------------------------------------------------------------------
__global__ __launch_bounds__(256) void ln_kernel(const float* __restrict__ x,
                                                 const float* __restrict__ gamma,
                                                 const float* __restrict__ beta,
                                                 __bf16* __restrict__ hout) {
    const int lane = threadIdx.x & 31;
    const int wave = threadIdx.x >> 5;
    const int token = blockIdx.x * 8 + wave;
    const float* xr = x + (size_t)token * EMBED;

    float v[24];
    float s = 0.f;
#pragma unroll
    for (int i = 0; i < 24; ++i) {
        v[i] = xr[lane + i * 32];
        s += v[i];
    }
#pragma unroll
    for (int off = 16; off; off >>= 1) s += __shfl_xor(s, off, 32);
    const float mu = s * (1.0f / EMBED);

    float ss = 0.f;
#pragma unroll
    for (int i = 0; i < 24; ++i) {
        const float d = v[i] - mu;
        ss += d * d;
    }
#pragma unroll
    for (int off = 16; off; off >>= 1) ss += __shfl_xor(ss, off, 32);
    const float rs = rsqrtf(ss * (1.0f / EMBED) + 1e-5f);

#pragma unroll
    for (int i = 0; i < 24; ++i) {
        const int d = lane + i * 32;
        hout[(size_t)token * EMBED + d] = f2bf((v[i] - mu) * rs * gamma[d] + beta[d]);
    }
}

// ---------------------------------------------------------------------------
// Kernel 2: cast w1 [2304x768] and w2 [768x768] fp32 -> bf16 (stay in L2).
// ---------------------------------------------------------------------------
__global__ __launch_bounds__(256) void cast_kernel(const float* __restrict__ w1,
                                                   const float* __restrict__ w2,
                                                   __bf16* __restrict__ w1b,
                                                   __bf16* __restrict__ w2b) {
    const int t = blockIdx.x * 256 + threadIdx.x;
    const int n1 = THREE_D * EMBED;
    if (t < n1)
        w1b[t] = f2bf(w1[t]);
    else
        w2b[t - n1] = f2bf(w2[t - n1]);
}

// ---------------------------------------------------------------------------
// Kernel 3: QKV GEMM. C[4096,2304] = h @ w1^T + b1.
// Block = 128 threads (4 waves, 2x2), block tile 128x128, wave tile 64x64
// (4x4 WMMA accumulators; 16 WMMAs per 16 b128 loads per K-step).
// Epilogue scatters Q,K as [B,H,L,Dh] and V transposed [B,H,Dh,L].
// ---------------------------------------------------------------------------
__global__ __launch_bounds__(128) void qkv_gemm_kernel(
    const __bf16* __restrict__ A, const __bf16* __restrict__ Bt,
    const float* __restrict__ bias, __bf16* __restrict__ Qb,
    __bf16* __restrict__ Kb, __bf16* __restrict__ Vt) {
    const int lane = threadIdx.x & 31;
    const int wave = threadIdx.x >> 5;
    const int m0 = blockIdx.x * 128 + (wave >> 1) * 64;
    const int n0 = blockIdx.y * 128 + (wave & 1) * 64;

    v8f acc[4][4] = {};
    for (int kb = 0; kb < EMBED; kb += 32) {
        if (kb + 32 < EMBED) {
            __builtin_prefetch(A + (size_t)(m0 + (lane & 15)) * EMBED + kb + 32, 0, 3);
            __builtin_prefetch(Bt + (size_t)(n0 + (lane & 15)) * EMBED + kb + 32, 0, 3);
        }
        v16bf af[4], bf_[4];
#pragma unroll
        for (int i = 0; i < 4; ++i)
            af[i] = load_frag_a(A + (size_t)(m0 + i * 16 + (lane & 15)) * EMBED + kb,
                                lane);
#pragma unroll
        for (int j = 0; j < 4; ++j)
            bf_[j] = load_frag_b(Bt + (size_t)(n0 + j * 16 + (lane & 15)) * EMBED + kb,
                                 lane);
#pragma unroll
        for (int i = 0; i < 4; ++i)
#pragma unroll
            for (int j = 0; j < 4; ++j) acc[i][j] = wmma_bf16(af[i], bf_[j], acc[i][j]);
    }

    const int hi8 = (lane & 16) ? 8 : 0;
#pragma unroll
    for (int i = 0; i < 4; ++i)
#pragma unroll
        for (int j = 0; j < 4; ++j)
#pragma unroll
            for (int r = 0; r < 8; ++r) {
                const int m = m0 + i * 16 + r + hi8;
                const int n = n0 + j * 16 + (lane & 15);
                const float val = acc[i][j][r] + bias[n];
                const int b = m >> 11, l = m & 2047;
                if (n < EMBED) {
                    const int h = n >> 6, dh = n & 63;
                    Qb[(((size_t)(b * HEADS + h) * SEQ) + l) * HDIM + dh] = f2bf(val);
                } else if (n < 2 * EMBED) {
                    const int n2 = n - EMBED, h = n2 >> 6, dh = n2 & 63;
                    Kb[(((size_t)(b * HEADS + h) * SEQ) + l) * HDIM + dh] = f2bf(val);
                } else {
                    const int n2 = n - 2 * EMBED, h = n2 >> 6, dh = n2 & 63;
                    Vt[(((size_t)(b * HEADS + h) * HDIM) + dh) * SEQ + l] = f2bf(val);
                }
            }
}

// ---------------------------------------------------------------------------
// Kernel 4: flash-style attention. Block = (b,h, 16-query tile), 4 waves.
// Per 64-key chunk: QK^T via WMMA -> LDS scores, online softmax, bf16 probs
// in LDS, O += P @ V via WMMA. With TDM: K and V^T chunks are DMA'd into
// double-buffered LDS by wave 0 (tensor_load_to_lds / s_wait_tensorcnt) so
// compute overlaps the next chunk's DMA and B-frags come from LDS.
// ---------------------------------------------------------------------------
__global__ __launch_bounds__(128) void attn_kernel(const __bf16* __restrict__ Qg,
                                                   const __bf16* __restrict__ Kg,
                                                   const __bf16* __restrict__ Vtg,
                                                   __bf16* __restrict__ Og) {
    const int lane = threadIdx.x & 31;
    const int wave = threadIdx.x >> 5;  // 0..3 -> Dh slice [wave*16, +16)
    const int tid = threadIdx.x;
    const int bh = blockIdx.x >> 7;
    const int qt = blockIdx.x & 127;
    const int qbase = qt * 16;
    const int b = bh / HEADS, h = bh % HEADS;

    const __bf16* Q = Qg + (size_t)bh * SEQ * HDIM;
    const __bf16* K = Kg + (size_t)bh * SEQ * HDIM;
    const __bf16* Vt = Vtg + (size_t)bh * HDIM * SEQ;

    __shared__ __align__(16) __bf16 Qs[16 * HDIM];
    __shared__ float S[16][64];
    __shared__ __align__(16) __bf16 P[16 * 64];
    __shared__ float mrow[16], lrow[16], arow[16];
#if USE_TDM
    __shared__ __align__(16) __bf16 Ks[2][64 * 64];  // [key][dh]
    __shared__ __align__(16) __bf16 Vs[2][64 * 64];  // [dh][key]
#endif

    // 16x64 bf16 Q tile = 2048B = 128 lanes x 16B
    ((uint4*)Qs)[tid] = ((const uint4*)(Q + (size_t)qbase * HDIM))[tid];
    if (tid < 16) {
        mrow[tid] = -1e30f;
        lrow[tid] = 0.f;
    }
    v8f o = {};

#if USE_TDM
    if (wave == 0) {  // prologue: DMA chunk 0 into buffer 0
        tdm_load_2d((unsigned)(size_t)(void*)&Ks[0][0], K, HDIM, 64, HDIM);
        tdm_load_2d((unsigned)(size_t)(void*)&Vs[0][0], Vt, 64, HDIM, SEQ);
    }
#endif
    __syncthreads();

    const int hi8 = (lane & 16) ? 8 : 0;
    const int qr = lane & 15;

    for (int c = 0; c < SEQ / 64; ++c) {
        const int kc = c * 64;
#if USE_TDM
        const int buf = c & 1;
        if (wave == 0) {
            if (c + 1 < SEQ / 64) {  // DMA next chunk into other buffer
                tdm_load_2d((unsigned)(size_t)(void*)&Ks[buf ^ 1][0],
                            K + (size_t)(kc + 64) * HDIM, HDIM, 64, HDIM);
                tdm_load_2d((unsigned)(size_t)(void*)&Vs[buf ^ 1][0], Vt + kc + 64,
                            64, HDIM, SEQ);
                __builtin_amdgcn_s_wait_tensorcnt(2);  // chunk c done
            } else {
                __builtin_amdgcn_s_wait_tensorcnt(0);
            }
        }
        __syncthreads();  // publish chunk c's LDS tiles to all waves
        const __bf16* Kc = &Ks[buf][0];   // rows: key (local), 64 dh contiguous
        const __bf16* Vc = &Vs[buf][0];   // rows: dh, 64 keys contiguous
        const int vstride = 64;
        const int krow0 = wave * 16;
#else
        const __bf16* Kc = K + (size_t)kc * HDIM;
        const __bf16* Vc = Vt + kc;
        const int vstride = SEQ;
        const int krow0 = wave * 16;
#endif
        // ---- phase A: S[16][w*16..] = Q(16x64) @ K^T chunk, scaled 1/8
        {
            v8f s = {};
            v16bf qa0 = load_frag_a(Qs + qr * HDIM + 0, lane);
            v16bf qa1 = load_frag_a(Qs + qr * HDIM + 32, lane);
            const int keyrow = krow0 + (lane & 15);
            v16bf kb0 = load_frag_b(Kc + (size_t)keyrow * HDIM + 0, lane);
            v16bf kb1 = load_frag_b(Kc + (size_t)keyrow * HDIM + 32, lane);
            s = wmma_bf16(qa0, kb0, s);
            s = wmma_bf16(qa1, kb1, s);
#pragma unroll
            for (int r = 0; r < 8; ++r)
                S[r + hi8][wave * 16 + (lane & 15)] = s[r] * 0.125f;
        }
        __syncthreads();

        // ---- phase B: online softmax; wave w owns rows 4w..4w+3
#pragma unroll
        for (int i = 0; i < 4; ++i) {
            const int row = wave * 4 + i;
            const float v0 = S[row][lane];
            const float v1 = S[row][lane + 32];
            float cm = fmaxf(v0, v1);
#pragma unroll
            for (int off = 16; off; off >>= 1) cm = fmaxf(cm, __shfl_xor(cm, off, 32));
            const float mo = mrow[row];
            const float mn = fmaxf(mo, cm);
            const float al = __expf(mo - mn);
            const float p0 = __expf(v0 - mn);
            const float p1 = __expf(v1 - mn);
            float ps = p0 + p1;
#pragma unroll
            for (int off = 16; off; off >>= 1) ps += __shfl_xor(ps, off, 32);
            P[row * 64 + lane] = f2bf(p0);
            P[row * 64 + lane + 32] = f2bf(p1);
            if (lane == 0) {
                mrow[row] = mn;
                lrow[row] = lrow[row] * al + ps;
                arow[row] = al;
            }
        }
        __syncthreads();

        // ---- phase C: rescale O, O += P(16x64) @ V chunk (this wave's Dh slice)
#pragma unroll
        for (int r = 0; r < 8; ++r) o[r] *= arow[r + hi8];
        {
            v16bf pa0 = load_frag_a(P + qr * 64 + 0, lane);
            v16bf pa1 = load_frag_a(P + qr * 64 + 32, lane);
            const int dhn = wave * 16 + (lane & 15);
            v16bf vb0 = load_frag_b(Vc + (size_t)dhn * vstride + 0, lane);
            v16bf vb1 = load_frag_b(Vc + (size_t)dhn * vstride + 32, lane);
            o = wmma_bf16(pa0, vb0, o);
            o = wmma_bf16(pa1, vb1, o);
        }
        __syncthreads();
    }

    // normalize and write attn output as bf16 [B, L, H*Dh]
#pragma unroll
    for (int r = 0; r < 8; ++r) {
        const int row = r + hi8;
        const int q = qbase + row;
        const float val = o[r] / lrow[row];
        Og[((size_t)(b * SEQ + q)) * EMBED + h * HDIM + wave * 16 + (lane & 15)] =
            f2bf(val);
    }
}

// ---------------------------------------------------------------------------
// Kernel 5: output projection + bias + residual. out = x + attn @ w2^T + b2.
// Same 64x64-per-wave WMMA tiling; fp32 epilogue to d_out.
// ---------------------------------------------------------------------------
__global__ __launch_bounds__(128) void proj_kernel(const __bf16* __restrict__ A,
                                                   const __bf16* __restrict__ Bt,
                                                   const float* __restrict__ bias,
                                                   const float* __restrict__ resid,
                                                   float* __restrict__ out) {
    const int lane = threadIdx.x & 31;
    const int wave = threadIdx.x >> 5;
    const int m0 = blockIdx.x * 128 + (wave >> 1) * 64;
    const int n0 = blockIdx.y * 128 + (wave & 1) * 64;

    v8f acc[4][4] = {};
    for (int kb = 0; kb < EMBED; kb += 32) {
        if (kb + 32 < EMBED) {
            __builtin_prefetch(A + (size_t)(m0 + (lane & 15)) * EMBED + kb + 32, 0, 3);
            __builtin_prefetch(Bt + (size_t)(n0 + (lane & 15)) * EMBED + kb + 32, 0, 3);
        }
        v16bf af[4], bf_[4];
#pragma unroll
        for (int i = 0; i < 4; ++i)
            af[i] = load_frag_a(A + (size_t)(m0 + i * 16 + (lane & 15)) * EMBED + kb,
                                lane);
#pragma unroll
        for (int j = 0; j < 4; ++j)
            bf_[j] = load_frag_b(Bt + (size_t)(n0 + j * 16 + (lane & 15)) * EMBED + kb,
                                 lane);
#pragma unroll
        for (int i = 0; i < 4; ++i)
#pragma unroll
            for (int j = 0; j < 4; ++j) acc[i][j] = wmma_bf16(af[i], bf_[j], acc[i][j]);
    }

    const int hi8 = (lane & 16) ? 8 : 0;
#pragma unroll
    for (int i = 0; i < 4; ++i)
#pragma unroll
        for (int j = 0; j < 4; ++j)
#pragma unroll
            for (int r = 0; r < 8; ++r) {
                const int m = m0 + i * 16 + r + hi8;
                const int n = n0 + j * 16 + (lane & 15);
                const size_t idx = (size_t)m * EMBED + n;
                out[idx] = acc[i][j][r] + bias[n] + resid[idx];
            }
}

// ---------------------------------------------------------------------------
// Workspace layout (bytes); all offsets 16B-aligned.
// ---------------------------------------------------------------------------
#define WS_H 0u         /* h bf16       4096x768   6291456 B */
#define WS_W1B 6291456u /* w1 bf16      2304x768   3538944 B */
#define WS_W2B 9830400u /* w2 bf16       768x768   1179648 B */
#define WS_Q 11010048u  /* Q  bf16 [B,H,L,Dh]      6291456 B */
#define WS_K 17301504u  /* K  bf16 [B,H,L,Dh]      6291456 B */
#define WS_VT 23592960u /* V^T bf16 [B,H,Dh,L]     6291456 B */
#define WS_O 29884416u  /* attn out bf16 [B,L,D]   6291456 B */

extern "C" void kernel_launch(void* const* d_in, const int* in_sizes, int n_in,
                              void* d_out, int out_size, void* d_ws, size_t ws_size,
                              hipStream_t stream) {
    (void)in_sizes;
    (void)n_in;
    (void)out_size;
    (void)ws_size;
    const float* x = (const float*)d_in[0];
    const float* gamma = (const float*)d_in[1];
    const float* beta = (const float*)d_in[2];
    const float* w1 = (const float*)d_in[3];
    const float* b1 = (const float*)d_in[4];
    const float* w2 = (const float*)d_in[5];
    const float* b2 = (const float*)d_in[6];
    float* out = (float*)d_out;

    char* ws = (char*)d_ws;
    __bf16* hb = (__bf16*)(ws + WS_H);
    __bf16* w1b = (__bf16*)(ws + WS_W1B);
    __bf16* w2b = (__bf16*)(ws + WS_W2B);
    __bf16* Qb = (__bf16*)(ws + WS_Q);
    __bf16* Kb = (__bf16*)(ws + WS_K);
    __bf16* Vt = (__bf16*)(ws + WS_VT);
    __bf16* Ob = (__bf16*)(ws + WS_O);

    // 1) LayerNorm -> bf16 h (4096 tokens, 8 waves/block)
    ln_kernel<<<MTOK / 8, 256, 0, stream>>>(x, gamma, beta, hb);
    // 2) weight cast ((2304+768)*768 elements)
    cast_kernel<<<(THREE_D * EMBED + EMBED * EMBED) / 256, 256, 0, stream>>>(w1, w2,
                                                                            w1b, w2b);
    // 3) QKV GEMM: M=4096, N=2304, block tile 128x128
    qkv_gemm_kernel<<<dim3(MTOK / 128, THREE_D / 128), 128, 0, stream>>>(hb, w1b, b1,
                                                                        Qb, Kb, Vt);
    // 4) attention: B*H*(L/16) blocks
    attn_kernel<<<BATCH * HEADS * (SEQ / 16), 128, 0, stream>>>(Qb, Kb, Vt, Ob);
    // 5) output projection + residual: N=768, block tile 128x128
    proj_kernel<<<dim3(MTOK / 128, EMBED / 128), 128, 0, stream>>>(Ob, w2b, b2, x, out);
}
